// Model_14456859918500
// MI455X (gfx1250) — matrix operand order repaired and checked by reading, not compiled, not run on hardware
//
#include <hip/hip_runtime.h>
#include <math.h>

// ---------------------------------------------------------------------------
// Types for CDNA5 WMMA (wave32): v_wmma_f32_16x16x32_bf16
// ---------------------------------------------------------------------------
typedef __attribute__((ext_vector_type(16))) __bf16 bf16x16;
typedef __attribute__((ext_vector_type(8)))  __bf16 bf16x8;
typedef __attribute__((ext_vector_type(8)))  float  v8f;

#define WMMA_BF16(a, b, c) \
    __builtin_amdgcn_wmma_f32_16x16x32_bf16(false, (a), false, (b), (short)0, (c), false, false)

__device__ __forceinline__ bf16x8 load_bf8(const __bf16* p) {
    union { uint4 u; bf16x8 v; } t;
    t.u = *reinterpret_cast<const uint4*>(p);
    return t.v;
}

__device__ __forceinline__ bf16x16 cat8(bf16x8 lo, bf16x8 hi) {
    bf16x16 r;
#pragma unroll
    for (int i = 0; i < 8; ++i) { r[i] = lo[i]; r[i + 8] = hi[i]; }
    return r;
}

// 16 contiguous bf16
__device__ __forceinline__ bf16x16 load_bf16c(const __bf16* p) {
    return cat8(load_bf8(p), load_bf8(p + 8));
}

// 16 contiguous fp32 -> bf16 fragment half
__device__ __forceinline__ bf16x16 load_f32x16_bf(const float* p) {
    bf16x16 r;
    const float4* q = reinterpret_cast<const float4*>(p);
#pragma unroll
    for (int i = 0; i < 4; ++i) {
        float4 f = q[i];
        r[4 * i + 0] = (__bf16)f.x; r[4 * i + 1] = (__bf16)f.y;
        r[4 * i + 2] = (__bf16)f.z; r[4 * i + 3] = (__bf16)f.w;
    }
    return r;
}

__device__ __forceinline__ float sigmoidf_(float x) { return 1.0f / (1.0f + __expf(-x)); }

// ---------------------------------------------------------------------------
// Problem constants
// ---------------------------------------------------------------------------
#define B_SZ   16
#define SEQ    48
#define EMB    256
#define HID    512
#define G4H    2048     // 4*H
#define ROWS   768      // B_SZ*SEQ
#define SLOT   122
#define INTENT 22
#define IDIM   128
#define SLOTN  62464    // SLOT*HID
#define INTN   2816     // INTENT*IDIM
#define HSK    1024     // 2*H

// ---------------------------------------------------------------------------
// 1) Embedding gather, fp32 -> bf16
// ---------------------------------------------------------------------------
__global__ void embed_kernel(const int* __restrict__ tok, const float* __restrict__ emb,
                             __bf16* __restrict__ x1b) {
    int row = blockIdx.x;                 // b*48 + t
    int e   = threadIdx.x;                // 0..255
    int id  = tok[row];
    x1b[(size_t)row * EMB + e] = (__bf16)emb[(size_t)id * EMB + e];
}

// ---------------------------------------------------------------------------
// 2) Transpose + convert: W fp32 (K x N, row-major) -> Wt bf16 (N x K, row-major)
//    so WMMA B-fragment loads become K-contiguous per lane.
// ---------------------------------------------------------------------------
__global__ void transpose_cvt_kernel(const float* __restrict__ W, __bf16* __restrict__ Wt,
                                     int K, int N) {
    __shared__ float tile[32][33];
    int n0 = blockIdx.x * 32, k0 = blockIdx.y * 32;
    int t = threadIdx.x;                  // 256
    int tn = t & 31, tk = t >> 5;         // 8 k-rows per pass
#pragma unroll
    for (int i = 0; i < 4; ++i) {
        int k = tk + i * 8;
        tile[k][tn] = W[(size_t)(k0 + k) * N + n0 + tn];
    }
    __syncthreads();
#pragma unroll
    for (int i = 0; i < 4; ++i) {
        int n = tk + i * 8;
        Wt[(size_t)(n0 + n) * K + k0 + tn] = (__bf16)tile[tn][n];
    }
}

// ---------------------------------------------------------------------------
// 3) Input projection: G_in[dir] = x1 @ wih.T + b   (M=768, K=256, N=2048)
//    wih is (2048,256) row-major => B[k][n]=wih[n*256+k] is K-contiguous. WMMA.
// ---------------------------------------------------------------------------
__global__ void gin_kernel(const __bf16* __restrict__ x1b,
                           const float* __restrict__ wih_f, const float* __restrict__ wih_b,
                           const float* __restrict__ b_f,   const float* __restrict__ b_b,
                           float* __restrict__ Gin) {
    int dir = blockIdx.z;
    const float* wih  = dir ? wih_b : wih_f;
    const float* bias = dir ? b_b   : b_f;
    int tid = threadIdx.x, wave = tid >> 5, lane = tid & 31;
    int mt = blockIdx.y;
    int Nbase = blockIdx.x * 512 + wave * 64;
    int lm = lane & 15;
    int alo = (lane < 16) ? 0 : 8;
    int bko = (lane < 16) ? 0 : 16;
    v8f acc[4] = {};
    const __bf16* ap = x1b + (size_t)(mt * 16 + lm) * EMB;
    for (int kc = 0; kc < EMB / 32; ++kc) {
        int kb = kc * 32;
        bf16x16 af = cat8(load_bf8(ap + kb + bko), load_bf8(ap + kb + bko + 8));
#pragma unroll
        for (int j = 0; j < 4; ++j) {
            const float* wp = wih + (size_t)(Nbase + j * 16 + lm) * EMB + kb + bko;
            bf16x16 bf = load_f32x16_bf(wp);
            acc[j] = WMMA_BF16(af, bf, acc[j]);
        }
    }
    int mofs = (lane < 16) ? 0 : 8;
    float* G = Gin + (size_t)dir * ROWS * G4H;
#pragma unroll
    for (int j = 0; j < 4; ++j) {
        int n = Nbase + j * 16 + lm;
        float bv = bias[n];
#pragma unroll
        for (int rr = 0; rr < 8; ++rr) {
            int m = mt * 16 + rr + mofs;
            G[(size_t)m * G4H + n] = acc[j][rr] + bv;
        }
    }
}

// ---------------------------------------------------------------------------
// 4) LSTM, one 1024-thread block per direction. Wave w owns hidden units
//    [w*16, w*16+16): accumulates all four gate tiles via WMMA from LDS h.
// ---------------------------------------------------------------------------
__global__ void __launch_bounds__(1024) lstm_kernel(const float* __restrict__ whh_f,
                                                    const float* __restrict__ whh_b,
                                                    const float* __restrict__ Gin,
                                                    __bf16* __restrict__ hseq) {
    int dir = blockIdx.x;
    const float* whh = dir ? whh_b : whh_f;
    const float* G   = Gin + (size_t)dir * ROWS * G4H;
    __shared__ __bf16 hbuf[B_SZ * HID];   // 16 KB
    int tid = threadIdx.x, wave = tid >> 5, lane = tid & 31;
    int hu = wave * 16;
    int lm = lane & 15;
    int alo = (lane < 16) ? 0 : 8;
    int bko = (lane < 16) ? 0 : 16;
    int mofs = (lane < 16) ? 0 : 8;
    for (int i = tid; i < B_SZ * HID; i += 1024) hbuf[i] = (__bf16)0.0f;
    v8f c = {};
    __syncthreads();
    for (int step = 0; step < SEQ; ++step) {
        int t = dir ? (SEQ - 1 - step) : step;
        v8f acc[4] = {};
        const __bf16* ap = hbuf + lm * HID;
        for (int kc = 0; kc < HID / 32; ++kc) {
            int kb = kc * 32;
            bf16x16 af = cat8(load_bf8(ap + kb + bko), load_bf8(ap + kb + bko + 8));
#pragma unroll
            for (int g = 0; g < 4; ++g) {
                const float* wp = whh + (size_t)(g * HID + hu + lm) * HID + kb + bko;
                bf16x16 bf = load_f32x16_bf(wp);
                acc[g] = WMMA_BF16(af, bf, acc[g]);
            }
        }
        __syncthreads();                  // all reads of hbuf done
#pragma unroll
        for (int rr = 0; rr < 8; ++rr) {
            int m = rr + mofs;            // batch index
            size_t grow = (size_t)(m * SEQ + t) * G4H;
            int n = hu + lm;
            float gi = acc[0][rr] + G[grow + n];
            float gf = acc[1][rr] + G[grow + HID + n];
            float gg = acc[2][rr] + G[grow + 2 * HID + n];
            float go = acc[3][rr] + G[grow + 3 * HID + n];
            float cn = sigmoidf_(gf) * c[rr] + sigmoidf_(gi) * tanhf(gg);
            c[rr] = cn;
            float h = sigmoidf_(go) * tanhf(cn);
            hbuf[m * HID + n] = (__bf16)h;
            hseq[(size_t)(m * SEQ + t) * HSK + dir * HID + n] = (__bf16)h;
        }
        __syncthreads();                  // writes visible before next step
    }
}

// ---------------------------------------------------------------------------
// 5) Generic WMMA GEMM with tanh epilogue: out = tanh(A @ Wt^T)
//    A: (M,K) bf16 row-major; Wt: (N,K) bf16 row-major (pre-transposed W).
//    Block: 256 threads = 8 waves, block tile M64 x N256, wave tile M32 x N64.
//    M-tail predication is wave-uniform so EXEC stays all-ones at every WMMA.
// ---------------------------------------------------------------------------
__global__ void gemm_tanh_kernel(const __bf16* __restrict__ A, const __bf16* __restrict__ Wt,
                                 __bf16* __restrict__ out, int M, int K, int N) {
    int tid = threadIdx.x, wave = tid >> 5, lane = tid & 31;
    int wm = wave >> 2, wn = wave & 3;
    int Mbase = blockIdx.y * 64 + wm * 32;
    int Nbase = blockIdx.x * 256 + wn * 64;
    bool v0 = (Mbase + 16) <= M;
    bool v1 = (Mbase + 32) <= M;
    int lm = lane & 15;
    int alo = (lane < 16) ? 0 : 8;
    int bko = (lane < 16) ? 0 : 16;
    v8f acc[2][4] = {};
    const __bf16* a0 = A + (size_t)(Mbase + lm) * K;
    const __bf16* a1 = A + (size_t)(Mbase + 16 + lm) * K;
    int KC = K >> 5;
    for (int kc = 0; kc < KC; ++kc) {
        int kb = kc << 5;
        bf16x16 af0 = {}, af1 = {};
        if (v0) af0 = cat8(load_bf8(a0 + kb + bko), load_bf8(a0 + kb + bko + 8));
        if (v1) af1 = cat8(load_bf8(a1 + kb + bko), load_bf8(a1 + kb + bko + 8));
        if (v0 && kc + 1 < KC) __builtin_prefetch(a0 + kb + 32, 0, 1);  // global_prefetch_b8
#pragma unroll
        for (int j = 0; j < 4; ++j) {
            const __bf16* bp = Wt + (size_t)(Nbase + j * 16 + lm) * K + kb + bko;
            bf16x16 bf = load_bf16c(bp);
            acc[0][j] = WMMA_BF16(af0, bf, acc[0][j]);
            acc[1][j] = WMMA_BF16(af1, bf, acc[1][j]);
        }
    }
    int mofs = (lane < 16) ? 0 : 8;
    for (int sub = 0; sub < 2; ++sub) {
        if (!(sub ? v1 : v0)) continue;
        int msub = Mbase + sub * 16;
#pragma unroll
        for (int j = 0; j < 4; ++j) {
            size_t ncol = (size_t)(Nbase + j * 16 + lm);
#pragma unroll
            for (int rr = 0; rr < 8; ++rr) {
                int m = msub + rr + mofs;
                out[(size_t)m * N + ncol] = (__bf16)tanhf(acc[sub][j][rr]);
            }
        }
    }
}

// ---------------------------------------------------------------------------
// Routing kernels (generic over S-dim, N-dim, D-dim)
// ---------------------------------------------------------------------------
__global__ void softmax_kernel(const float* __restrict__ Bm, float* __restrict__ C, int N) {
    int row = blockIdx.x, t = threadIdx.x;       // block 128
    __shared__ float red[128];
    float v = (t < N) ? Bm[(size_t)row * N + t] : -3.4e38f;
    red[t] = v; __syncthreads();
    for (int s = 64; s > 0; s >>= 1) { if (t < s) red[t] = fmaxf(red[t], red[t + s]); __syncthreads(); }
    float mx = red[0]; __syncthreads();
    float e = (t < N) ? __expf(v - mx) : 0.0f;
    red[t] = e; __syncthreads();
    for (int s = 64; s > 0; s >>= 1) { if (t < s) red[t] += red[t + s]; __syncthreads(); }
    float inv = 1.0f / red[0];
    if (t < N) C[(size_t)row * N + t] = e * inv;
}

// S[b,n,:] = sum_s C[b,s,n]*uhat[b,s,n,:];  V = S / sqrt(|S|^2 + 1e-7)
__global__ void sv_kernel(const float* __restrict__ C, const __bf16* __restrict__ uhat,
                          float* __restrict__ V, int S, int Nn, int D) {
    int n = blockIdx.x, b = blockIdx.y, t = threadIdx.x;   // block 128
    int nd = D >> 7;                                       // 4 (D=512) or 1 (D=128)
    float acc[4] = {0.f, 0.f, 0.f, 0.f};
    for (int s = 0; s < S; ++s) {
        float cc = C[((size_t)(b * S + s)) * Nn + n];
        const __bf16* up = uhat + (((size_t)(b * S + s)) * Nn + n) * D;
        for (int i = 0; i < nd; ++i) acc[i] += cc * (float)up[t + (i << 7)];
    }
    float ss = 0.f;
    for (int i = 0; i < nd; ++i) ss += acc[i] * acc[i];
    __shared__ float red[128];
    red[t] = ss; __syncthreads();
    for (int s2 = 64; s2 > 0; s2 >>= 1) { if (t < s2) red[t] += red[t + s2]; __syncthreads(); }
    float scale = 1.0f / sqrtf(red[0] + 1e-7f);
    float* vp = V + ((size_t)(b * Nn + n)) * D;
    for (int i = 0; i < nd; ++i) vp[t + (i << 7)] = acc[i] * scale;
}

// B[b,s,n] += dot(uhat[b,s,n,:],V[b,n,:]) + coef*dot(uhat[b,s,n,:], r[b,:])
__global__ void agree_kernel(const __bf16* __restrict__ uhat, const float* __restrict__ V,
                             float* __restrict__ Bm, const float* __restrict__ r,
                             int S, int Nn, int D, float coef) {
    int idx = blockIdx.x;
    int n = idx % Nn; int bs = idx / Nn; int s = bs % S; int b = bs / S;
    int t = threadIdx.x;                                   // block 64
    const __bf16* up = uhat + (((size_t)(b * S + s)) * Nn + n) * D;
    const float* vp = V + ((size_t)(b * Nn + n)) * D;
    const float* rp = r ? (r + (size_t)b * HID) : nullptr;
    float p1 = 0.f, p2 = 0.f;
    for (int d = t; d < D; d += 64) {
        float u = (float)up[d];
        p1 += u * vp[d];
        if (rp) p2 += u * rp[d];
    }
    __shared__ float r1[64], r2[64];
    r1[t] = p1; r2[t] = p2; __syncthreads();
    for (int s2 = 32; s2 > 0; s2 >>= 1) {
        if (t < s2) { r1[t] += r1[t + s2]; r2[t] += r2[t + s2]; }
        __syncthreads();
    }
    if (t == 0) Bm[idx] += r1[0] + coef * r2[0];
}

// r[b,u] = sum_d w_rr[u,d] * V_int[b, argmax_n||V_int[b,n]||, d]
__global__ void argmax_r_kernel(const float* __restrict__ Vint, const float* __restrict__ w_rr,
                                float* __restrict__ r) {
    int b = blockIdx.x, t = threadIdx.x;                   // block 512
    __shared__ float norms[INTENT];
    __shared__ int best;
    if (t < INTENT) {
        const float* vp = Vint + ((size_t)(b * INTENT + t)) * IDIM;
        float ss = 0.f;
        for (int d = 0; d < IDIM; ++d) { float v = vp[d]; ss += v * v; }
        norms[t] = ss;
    }
    __syncthreads();
    if (t == 0) {
        int bi = 0; float bv = norms[0];
        for (int i = 1; i < INTENT; ++i) if (norms[i] > bv) { bv = norms[i]; bi = i; }
        best = bi;
    }
    __syncthreads();
    const float* vp = Vint + ((size_t)(b * INTENT + best)) * IDIM;
    const float* wp = w_rr + (size_t)t * IDIM;
    float acc = 0.f;
    for (int d = 0; d < IDIM; ++d) acc += wp[d] * vp[d];
    r[(size_t)b * HID + t] = acc;
}

// ---------------------------------------------------------------------------
// Small utility kernels
// ---------------------------------------------------------------------------
__global__ void zero_f32_kernel(float* p, int n) {
    int i = blockIdx.x * 256 + threadIdx.x; if (i < n) p[i] = 0.0f;
}
__global__ void cvt_bf16_kernel(const float* in, __bf16* out, int n) {
    int i = blockIdx.x * 256 + threadIdx.x; if (i < n) out[i] = (__bf16)in[i];
}
__global__ void copy_f32_kernel(const float* in, float* out, int n) {
    int i = blockIdx.x * 256 + threadIdx.x; if (i < n) out[i] = in[i];
}

// ---------------------------------------------------------------------------
// Host orchestration (all launches on `stream`; workspace carved from d_ws)
// ---------------------------------------------------------------------------
extern "C" void kernel_launch(void* const* d_in, const int* in_sizes, int n_in,
                              void* d_out, int out_size, void* d_ws, size_t ws_size,
                              hipStream_t stream) {
    const int*   tokens   = (const int*)  d_in[0];
    const float* emb      = (const float*)d_in[1];
    const float* wih_f    = (const float*)d_in[2];
    const float* whh_f    = (const float*)d_in[3];
    const float* b_f      = (const float*)d_in[4];
    const float* wih_b    = (const float*)d_in[5];
    const float* whh_b    = (const float*)d_in[6];
    const float* b_b      = (const float*)d_in[7];
    const float* slot_W   = (const float*)d_in[8];
    const float* w_rr     = (const float*)d_in[9];
    const float* intent_W = (const float*)d_in[10];
    float* out = (float*)d_out;

    char* ws = (char*)d_ws;
    size_t off = 0;
    auto alloc = [&](size_t bytes) -> char* {
        char* p = ws + off;
        off = (off + bytes + 255) & ~(size_t)255;
        return p;
    };
    __bf16* x1b   = (__bf16*)alloc((size_t)ROWS * EMB * 2);
    __bf16* WtS   = (__bf16*)alloc((size_t)SLOTN * HSK * 2);     // slot_W^T bf16, 122 MB
    __bf16* WtI   = (__bf16*)alloc((size_t)INTN * HID * 2);
    float*  Gin   = (float*) alloc((size_t)2 * ROWS * G4H * 4);
    __bf16* hseqb = (__bf16*)alloc((size_t)ROWS * HSK * 2);
    __bf16* uhS   = (__bf16*)alloc((size_t)ROWS * SLOTN * 2);    // 91.5 MB
    __bf16* uhI   = (__bf16*)alloc((size_t)B_SZ * SLOT * INTN * 2);
    float*  Vs    = (float*) alloc((size_t)B_SZ * SLOT * HID * 4);
    __bf16* Vsb   = (__bf16*)alloc((size_t)B_SZ * SLOT * HID * 2);
    float*  Vi    = (float*) alloc((size_t)B_SZ * INTENT * IDIM * 4);
    float*  Ctmp  = (float*) alloc((size_t)ROWS * SLOT * 4);
    float*  Bs    = (float*) alloc((size_t)ROWS * SLOT * 4);
    float*  Bi    = (float*) alloc((size_t)B_SZ * SLOT * INTENT * 4);
    float*  Bs2   = (float*) alloc((size_t)ROWS * SLOT * 4);
    float*  Rr    = (float*) alloc((size_t)B_SZ * HID * 4);

    const int nBs  = ROWS * SLOT;           // 93696
    const int nBi  = B_SZ * SLOT * INTENT;  // 42944
    float* outVint   = out + nBs;
    float* outCslot2 = outVint + B_SZ * INTENT * IDIM;
    float* outCint   = outCslot2 + 3 * nBs;

    // Stage 1: embedding, weight transposes, input projections, BiLSTM
    embed_kernel<<<ROWS, EMB, 0, stream>>>(tokens, emb, x1b);
    transpose_cvt_kernel<<<dim3(SLOTN / 32, HSK / 32), 256, 0, stream>>>(slot_W, WtS, HSK, SLOTN);
    transpose_cvt_kernel<<<dim3(INTN / 32, HID / 32), 256, 0, stream>>>(intent_W, WtI, HID, INTN);
    gin_kernel<<<dim3(4, ROWS / 16, 2), 256, 0, stream>>>(x1b, wih_f, wih_b, b_f, b_b, Gin);
    lstm_kernel<<<2, 1024, 0, stream>>>(whh_f, whh_b, Gin, hseqb);

    // Stage 2: uhat_s = tanh(hseq @ slot_W)    M=768 K=1024 N=62464
    gemm_tanh_kernel<<<dim3(SLOTN / 256, ROWS / 64), 256, 0, stream>>>(hseqb, WtS, uhS, ROWS, HSK, SLOTN);

    // Stage 3: slot routing phase 1 (V_slot only)
    zero_f32_kernel<<<(nBs + 255) / 256, 256, 0, stream>>>(Bs, nBs);
    for (int it = 0; it < 3; ++it) {
        softmax_kernel<<<ROWS, 128, 0, stream>>>(Bs, Ctmp, SLOT);
        sv_kernel<<<dim3(SLOT, B_SZ), 128, 0, stream>>>(Ctmp, uhS, Vs, SEQ, SLOT, HID);
        if (it < 2)
            agree_kernel<<<nBs, 64, 0, stream>>>(uhS, Vs, Bs, nullptr, SEQ, SLOT, HID, 0.0f);
    }

    // Stage 4: uhat_i = tanh(V_slot @ intent_W)   M=1952 K=512 N=2816
    cvt_bf16_kernel<<<(B_SZ * SLOT * HID + 255) / 256, 256, 0, stream>>>(Vs, Vsb, B_SZ * SLOT * HID);
    gemm_tanh_kernel<<<dim3(INTN / 256, (B_SZ * SLOT + 63) / 64), 256, 0, stream>>>(
        Vsb, WtI, uhI, B_SZ * SLOT, HID, INTN);

    // Stage 5: intent routing (C_int to d_out each iter; V_int to d_out at final iter)
    zero_f32_kernel<<<(nBi + 255) / 256, 256, 0, stream>>>(Bi, nBi);
    for (int it = 0; it < 3; ++it) {
        float* Cout = outCint + (size_t)it * nBi;
        softmax_kernel<<<B_SZ * SLOT, 128, 0, stream>>>(Bi, Cout, INTENT);
        sv_kernel<<<dim3(INTENT, B_SZ), 128, 0, stream>>>(Cout, uhI, (it == 2) ? outVint : Vi,
                                                          SLOT, INTENT, IDIM);
        if (it < 2)
            agree_kernel<<<nBi, 64, 0, stream>>>(uhI, Vi, Bi, nullptr, SLOT, INTENT, IDIM, 0.0f);
    }

    // Stage 6: ihat -> r[b] = w_rr @ ihat[b]  (folds the 'bsnu,ud ; bsnd,bd' chain)
    argmax_r_kernel<<<B_SZ, HID, 0, stream>>>(outVint, w_rr, Rr);

    // Stage 7: slot routing phase 2 with 0.1 * (uhat . r) term; emit C_slot2 and slot_p
    zero_f32_kernel<<<(nBs + 255) / 256, 256, 0, stream>>>(Bs2, nBs);
    for (int it = 0; it < 3; ++it) {
        float* Cout = outCslot2 + (size_t)it * nBs;
        softmax_kernel<<<ROWS, 128, 0, stream>>>(Bs2, Cout, SLOT);
        if (it < 2) {
            sv_kernel<<<dim3(SLOT, B_SZ), 128, 0, stream>>>(Cout, uhS, Vs, SEQ, SLOT, HID);
            agree_kernel<<<nBs, 64, 0, stream>>>(uhS, Vs, Bs2, Rr, SEQ, SLOT, HID, 0.1f);
        }
    }
    // slot_p = B_logits after 2 updates
    copy_f32_kernel<<<(nBs + 255) / 256, 256, 0, stream>>>(Bs2, out, nBs);
}
